// ESRNN_model_26912265077444
// MI455X (gfx1250) — matrix operand order repaired and checked by reading, not compiled, not run on hardware
//
#include <hip/hip_runtime.h>
#include <cstddef>
#include <cstdint>

// ---------------------------------------------------------------------------
// Problem constants (from reference): S=12, IS=24, OS=12, H=512, C=6
// B=512, T=240, W=217, num_out=205, dilations (1,2),(4,8)
// ---------------------------------------------------------------------------

typedef __attribute__((ext_vector_type(16))) __bf16 v16bf;
typedef __attribute__((ext_vector_type(8)))  float  v8f;
typedef __attribute__((ext_vector_type(4)))  unsigned int v4u;

union V16 {
  v16bf v;
  v4u   q[2];
  unsigned short u[16];
};

__device__ __forceinline__ unsigned short f2bf(float f) {
  unsigned int u = __float_as_uint(f);
  unsigned int r = u + 0x7FFFu + ((u >> 16) & 1u);   // round-to-nearest-even
  return (unsigned short)(r >> 16);
}
__device__ __forceinline__ float bf2f(unsigned short h) {
  return __uint_as_float(((unsigned int)h) << 16);
}
__device__ __forceinline__ float sigm(float x) { return 1.0f / (1.0f + expf(-x)); }

__device__ __forceinline__ v8f wmma_bf16(v8f c, v16bf a, v16bf b) {
  // D = A(16x32 bf16) x B(32x16 bf16) + C(16x16 f32)
  return __builtin_amdgcn_wmma_f32_16x16x32_bf16(false, a, false, b, (short)0, c, false, false);
}

// ---------------------------------------------------------------------------
// Exponential-smoothing prep: one thread per series (sequential T=240 scan).
// Produces levels (512x240), seasonalities (512x252), per-series loss partial.
// ---------------------------------------------------------------------------
__global__ void es_prep(const float* __restrict__ train, const float* __restrict__ lvc,
                        const float* __restrict__ ssc, const float* __restrict__ sp,
                        const int* __restrict__ idxs, float* __restrict__ levels,
                        float* __restrict__ seas, float* __restrict__ lpart) {
  int b = blockIdx.x * blockDim.x + threadIdx.x;
  if (b >= 512) return;
  int idx = idxs[b];
  float alpha = sigm(lvc[idx]);
  float gamma = sigm(ssc[idx]);
  float s0[13];
  for (int j = 0; j < 12; ++j) s0[j] = expf(sp[idx * 12 + j]);
  s0[12] = s0[0];
  for (int j = 0; j < 13; ++j) seas[b * 252 + j] = s0[j];
  float buf[12];
  for (int j = 0; j < 12; ++j) buf[j] = s0[j + 1];
  int head = 0;
  float lev = train[b * 240] / s0[0];
  levels[b * 240] = lev;
  const float seas1 = s0[1];                 // constant, per reference
  float prevlev = lev, prevld = 0.0f, acc = 0.0f;
  for (int t = 1; t < 240; ++t) {
    float x = train[b * 240 + t];
    lev = alpha * (x / seas1) + (1.0f - alpha) * prevlev;
    levels[b * 240 + t] = lev;
    float ld = logf(lev / prevlev);
    if (t >= 2) { float dd = ld - prevld; acc += dd * dd; }
    prevld = ld; prevlev = lev;
    float nw = gamma * (x / lev) + (1.0f - gamma) * buf[head];
    seas[b * 252 + 12 + t] = nw;             // seasonalities col 13+(t-1)
    buf[head] = nw;
    head = (head + 1 == 12) ? 0 : head + 1;
  }
  lpart[b] = acc;
}

__global__ void loss_reduce(const float* __restrict__ lpart, float* __restrict__ outv) {
  __shared__ float sm[512];
  int t = threadIdx.x;
  sm[t] = lpart[t];
  __syncthreads();
  for (int s = 256; s > 0; s >>= 1) { if (t < s) sm[t] += sm[t + s]; __syncthreads(); }
  if (t == 0) *outv = sm[0] / (238.0f * 512.0f);
}

// ---------------------------------------------------------------------------
// Build deseasonalized/normalized LSTM input (217,512,32) bf16, K padded 30->32
// cols 0..23 = windowed train, 24..29 = info_cat, 30..31 = 0
// ---------------------------------------------------------------------------
__global__ void build_inputs(const float* __restrict__ train, const float* __restrict__ seas,
                             const float* __restrict__ levels, const float* __restrict__ info,
                             unsigned short* __restrict__ x0) {
  int id = blockIdx.x * blockDim.x + threadIdx.x;
  if (id >= 217 * 512 * 32) return;
  int col = id & 31;
  int b = (id >> 5) & 511;
  int w = id >> 14;
  float v;
  if (col < 24) {
    int t = w + col;
    v = train[b * 240 + t] / seas[b * 252 + t] / levels[b * 240 + 23 + w];
  } else if (col < 30) {
    v = info[b * 6 + (col - 24)];
  } else {
    v = 0.0f;
  }
  x0[id] = f2bf(v);
}

// network_act = output_list (205,512,12)
__global__ void network_act_k(const float* __restrict__ train, const float* __restrict__ seas,
                              const float* __restrict__ levels, float* __restrict__ na) {
  int id = blockIdx.x * blockDim.x + threadIdx.x;
  if (id >= 205 * 512 * 12) return;
  int j = id % 12;
  int b = (id / 12) % 512;
  int w = id / 6144;
  int t = 24 + w + j;
  na[id] = train[b * 240 + t] / seas[b * 252 + t] / levels[b * 240 + 23 + w];
}

// hold_out_act = val ; hold_out_act_deseas_norm = val / seas[:,-12:] / levels[:,-1]
__global__ void holdout_k(const float* __restrict__ val, const float* __restrict__ seas,
                          const float* __restrict__ levels, float* __restrict__ ha,
                          float* __restrict__ hadn) {
  int id = blockIdx.x * blockDim.x + threadIdx.x;
  if (id >= 512 * 12) return;
  int b = id / 12, j = id % 12;
  float v = val[id];
  ha[id] = v;
  hadn[id] = v / seas[b * 252 + 240 + j] / levels[b * 240 + 239];
}

// ---------------------------------------------------------------------------
// Pack weight W (N rows x K cols, f32) into WMMA B-operand tiles (bf16):
// dst[((kt*Ntiles+nt)*32 + lane)*16 + e] = W[nt*16+(lane&15)][kt*32 + 8*(lane>>4) + (e&7) + 16*(e>>3)]
// so the GEMM does one contiguous 32B load per lane per K-tile.
// ---------------------------------------------------------------------------
__global__ void pack_w(const float* __restrict__ W, unsigned short* __restrict__ dst,
                       int N, int K, int Ntiles, int Ktiles) {
  int id = blockIdx.x * blockDim.x + threadIdx.x;
  if (id >= Ktiles * Ntiles * 512) return;
  int e = id & 15;
  int lane = (id >> 4) & 31;
  int nt = (id >> 9) % Ntiles;
  int kt = (id >> 9) / Ntiles;
  int n = nt * 16 + (lane & 15);
  int kk = kt * 32 + ((lane >> 4) << 3) + (e & 7) + ((e >> 3) << 4);
  float v = (n < N && kk < K) ? W[(size_t)n * K + kk] : 0.0f;
  dst[id] = f2bf(v);
}

// ---------------------------------------------------------------------------
// Fused per-timestep LSTM: gate GEMM (bf16 WMMA) + cell update in registers.
// Each wave owns 4 accumulators = the i,f,g,o gate tiles for the SAME 16
// hidden columns (N-tiles ntg, ntg+32, ntg+64, ntg+96 of the 2048-wide gate
// matrix). After the K-loops, accX[rr] holds the pre-activation of gate X at
// (row m, hidden col hc), so sigmoid/tanh/cell math happens in registers and
// the 33MB fp32 gates round-trip is eliminated. Writes: cstate f32,
// hstate bf16 (next-step A operand), aout bf16 (layer output, valid t only).
// 4 independent accumulators share one A fragment -> no WMMA RAW hazard NOPs.
// Grid (R/16, 4), 256 threads (8 waves x 4 hc-tile groups = 32 = 512/16).
// ---------------------------------------------------------------------------
__global__ __launch_bounds__(256) void lstm_step_fused(
    const unsigned short* __restrict__ xsrc, int ldx, int kxTiles,
    const unsigned short* __restrict__ wx,
    const unsigned short* __restrict__ wh,
    const float* __restrict__ bias,
    float* __restrict__ cstate, unsigned short* __restrict__ hstate,
    unsigned short* __restrict__ aout,
    int step, int d) {
  constexpr int NT = 128;                 // gate matrix: 2048/16 N-tiles
  constexpr int GATE = 32 * 512;          // pack offset between gate blocks (nt += 32)
  const int lane = threadIdx.x & 31;
  const int wave = threadIdx.x >> 5;
  const int ntg = blockIdx.y * 8 + wave;  // hidden-column tile 0..31
  const int mt = blockIdx.x;
  const int half = lane >> 4;
  const int mloc = lane & 15;
  const int r = mt * 16 + mloc;           // row within step batch
  const int jj = r >> 9;                  // dilation slot (uniform per tile)
  const int bb = r & 511;
  const int t = step * d + jj;            // original time index

  v8f accI = {}, accF = {}, accG = {}, accO = {};

  if (t < 217) {                          // X part (skipped for padding -> zero)
    const unsigned short* rowp = xsrc + ((size_t)t * 512 + bb) * (size_t)ldx;
    for (int kt = 0; kt < kxTiles; ++kt) {
      V16 a;
      const int kb = kt * 32 + half * 8;  // A layout: K = kb..kb+7, kb+16..kb+23
      a.q[0] = *(const v4u*)(rowp + kb);
      a.q[1] = *(const v4u*)(rowp + kb + 16);
      const unsigned short* bp = wx + (((size_t)kt * NT + ntg) * 32 + lane) * 16;
      V16 w0, w1, w2, w3;
      w0.q[0] = *(const v4u*)(bp);             w0.q[1] = *(const v4u*)(bp + 8);
      w1.q[0] = *(const v4u*)(bp + GATE);      w1.q[1] = *(const v4u*)(bp + GATE + 8);
      w2.q[0] = *(const v4u*)(bp + 2 * GATE);  w2.q[1] = *(const v4u*)(bp + 2 * GATE + 8);
      w3.q[0] = *(const v4u*)(bp + 3 * GATE);  w3.q[1] = *(const v4u*)(bp + 3 * GATE + 8);
      accI = wmma_bf16(accI, a.v, w0.v);
      accF = wmma_bf16(accF, a.v, w1.v);
      accG = wmma_bf16(accG, a.v, w2.v);
      accO = wmma_bf16(accO, a.v, w3.v);
    }
  }
  {                                       // H part (recurrent)
    const unsigned short* rowp = hstate + (size_t)r * 512;
#pragma unroll 4
    for (int kt = 0; kt < 16; ++kt) {
      V16 a;
      const int kb = kt * 32 + half * 8;
      a.q[0] = *(const v4u*)(rowp + kb);
      a.q[1] = *(const v4u*)(rowp + kb + 16);
      const unsigned short* bp = wh + (((size_t)kt * NT + ntg) * 32 + lane) * 16;
      V16 w0, w1, w2, w3;
      w0.q[0] = *(const v4u*)(bp);             w0.q[1] = *(const v4u*)(bp + 8);
      w1.q[0] = *(const v4u*)(bp + GATE);      w1.q[1] = *(const v4u*)(bp + GATE + 8);
      w2.q[0] = *(const v4u*)(bp + 2 * GATE);  w2.q[1] = *(const v4u*)(bp + 2 * GATE + 8);
      w3.q[0] = *(const v4u*)(bp + 3 * GATE);  w3.q[1] = *(const v4u*)(bp + 3 * GATE + 8);
      accI = wmma_bf16(accI, a.v, w0.v);
      accF = wmma_bf16(accF, a.v, w1.v);
      accG = wmma_bf16(accG, a.v, w2.v);
      accO = wmma_bf16(accO, a.v, w3.v);
    }
  }

  // ---- fused LSTM cell epilogue (C/D layout: n = lane&15, m = rr + 8*half) ----
  const int nl = lane & 15;
  const int hc = ntg * 16 + nl;           // hidden column 0..511
  const float bI = bias[hc];
  const float bF = bias[512 + hc];
  const float bG = bias[1024 + hc];
  const float bO = bias[1536 + hc];
  const int m0 = mt * 16 + half * 8;
  const bool valid = (t < 217);
#pragma unroll
  for (int rr = 0; rr < 8; ++rr) {
    const int m = m0 + rr;
    const size_t ix = (size_t)m * 512 + hc;
    float ip = accI[rr] + bI;
    float fp = accF[rr] + bF;
    float gp = accG[rr] + bG;
    float op = accO[rr] + bO;
    float cn = sigm(fp) * cstate[ix] + sigm(ip) * tanhf(gp);
    float hn = sigm(op) * tanhf(cn);
    cstate[ix] = cn;
    unsigned short hb = f2bf(hn);
    hstate[ix] = hb;
    if (valid) aout[((size_t)t * 512 + (m & 511)) * 512 + hc] = hb;
  }
}

// residual: o = a + b (bf16)
__global__ void resadd(const unsigned short* __restrict__ a, const unsigned short* __restrict__ b,
                       unsigned short* __restrict__ o, size_t n) {
  size_t i = (size_t)blockIdx.x * blockDim.x + threadIdx.x;
  if (i >= n) return;
  o[i] = f2bf(bf2f(a[i]) + bf2f(b[i]));
}

// head: z = tanh(h @ nl_W^T + nl_b), M=217*512, N=K=512, bf16 WMMA.
// 4-accumulator strip scheme; 8 waves cover all 32 N-tiles; grid (6944,1).
__global__ __launch_bounds__(256) void head_gemm(
    unsigned short* __restrict__ zout, const unsigned short* __restrict__ ain,
    const unsigned short* __restrict__ wp, const float* __restrict__ bias) {
  constexpr int NT = 32;
  const int lane = threadIdx.x & 31;
  const int wave = threadIdx.x >> 5;
  const int ntBase = wave * 4;
  const int mt = blockIdx.x;
  const int half = lane >> 4;
  const int mloc = lane & 15;
  const unsigned short* rowp = ain + (size_t)(mt * 16 + mloc) * 512;
  v8f acc0 = {}, acc1 = {}, acc2 = {}, acc3 = {};
#pragma unroll 4
  for (int kt = 0; kt < 16; ++kt) {
    V16 a;
    const int kb = kt * 32 + half * 8;
    a.q[0] = *(const v4u*)(rowp + kb);
    a.q[1] = *(const v4u*)(rowp + kb + 16);
    const unsigned short* bp = wp + (((size_t)kt * NT + ntBase) * 32 + lane) * 16;
    V16 w0, w1, w2, w3;
    w0.q[0] = *(const v4u*)(bp);            w0.q[1] = *(const v4u*)(bp + 8);
    w1.q[0] = *(const v4u*)(bp + 512);      w1.q[1] = *(const v4u*)(bp + 520);
    w2.q[0] = *(const v4u*)(bp + 1024);     w2.q[1] = *(const v4u*)(bp + 1032);
    w3.q[0] = *(const v4u*)(bp + 1536);     w3.q[1] = *(const v4u*)(bp + 1544);
    acc0 = wmma_bf16(acc0, a.v, w0.v);
    acc1 = wmma_bf16(acc1, a.v, w1.v);
    acc2 = wmma_bf16(acc2, a.v, w2.v);
    acc3 = wmma_bf16(acc3, a.v, w3.v);
  }
  const int nl = lane & 15;
#pragma unroll
  for (int rr = 0; rr < 8; ++rr) {
    int m = mt * 16 + half * 8 + rr;
    int c0 = ntBase * 16 + nl;
    zout[(size_t)m * 512 + c0]      = f2bf(tanhf(acc0[rr] + bias[c0]));
    zout[(size_t)m * 512 + c0 + 16] = f2bf(tanhf(acc1[rr] + bias[c0 + 16]));
    zout[(size_t)m * 512 + c0 + 32] = f2bf(tanhf(acc2[rr] + bias[c0 + 32]));
    zout[(size_t)m * 512 + c0 + 48] = f2bf(tanhf(acc3[rr] + bias[c0 + 48]));
  }
}

// final projection: out_full(m,12) = z(m,512) @ sc_W^T + sc_b (N=12: plain dot)
__global__ void sc_head(const unsigned short* __restrict__ z, const float* __restrict__ scW,
                        const float* __restrict__ scb, float* __restrict__ of) {
  int id = blockIdx.x * blockDim.x + threadIdx.x;
  if (id >= 217 * 512 * 12) return;
  int j = id % 12;
  size_t m = (size_t)(id / 12);
  const unsigned short* zr = z + m * 512;
  const float* wr = scW + j * 512;
  float acc = scb[j];
  for (int k = 0; k < 512; ++k) acc += bf2f(zr[k]) * wr[k];
  of[id] = acc;
}

// scatter outputs: predictions = out_full[:205] (causality), output_non_train,
// hold_out_pred = relu(out_full[-1] * seas[:,-12:] * levels[:,-1])
__global__ void finalize_k(const float* __restrict__ of, const float* __restrict__ seas,
                           const float* __restrict__ levels, float* __restrict__ pred,
                           float* __restrict__ ont, float* __restrict__ hop) {
  int id = blockIdx.x * blockDim.x + threadIdx.x;
  if (id >= 217 * 512 * 12) return;
  int j = id % 12;
  int b = (id / 12) % 512;
  int w = id / 6144;
  float v = of[id];
  ont[id] = v;
  if (w < 205) pred[id] = v;
  if (w == 216) {
    float rs = v * seas[b * 252 + 240 + j] * levels[b * 240 + 239];
    hop[b * 12 + j] = rs > 0.0f ? rs : 0.0f;
  }
}

// ---------------------------------------------------------------------------
// Host driver
// ---------------------------------------------------------------------------
extern "C" void kernel_launch(void* const* d_in, const int* in_sizes, int n_in,
                              void* d_out, int out_size, void* d_ws, size_t ws_size,
                              hipStream_t stream) {
  (void)in_sizes; (void)n_in; (void)out_size; (void)ws_size;
  const float* train = (const float*)d_in[0];
  const float* val   = (const float*)d_in[1];
  const float* info  = (const float*)d_in[2];
  const float* lvc   = (const float*)d_in[3];
  const float* ssc   = (const float*)d_in[4];
  const float* sp    = (const float*)d_in[5];
  const float* Wih[4]; const float* Whh[4]; const float* bi[4];
  for (int l = 0; l < 4; ++l) {
    Wih[l] = (const float*)d_in[6 + 3 * l];
    Whh[l] = (const float*)d_in[7 + 3 * l];
    bi[l]  = (const float*)d_in[8 + 3 * l];
  }
  const float* nlW = (const float*)d_in[18];
  const float* nlb = (const float*)d_in[19];
  const float* scW = (const float*)d_in[20];
  const float* scb = (const float*)d_in[21];
  const int*  idxs = (const int*)d_in[22];
  float* out = (float*)d_out;

  // output offsets (flat, return order)
  const size_t PRED = 0;              // (205,512,12)
  const size_t NA   = 1259520;        // (205,512,12)
  const size_t HOP  = 2519040;        // (512,12)
  const size_t ONT  = 2525184;        // (217,512,12)
  const size_t HA   = 3858432;        // (512,12)
  const size_t HADN = 3864576;        // (512,12)
  const size_t LOSS = 3870720;        // scalar

  // workspace carve (bump allocator, 256B aligned)
  char* base = (char*)d_ws;
  size_t off = 0;
  auto alloc = [&](size_t bytes) -> void* {
    off = (off + 255) & ~(size_t)255;
    void* p = base + off;
    off += bytes;
    return p;
  };
  float* levels = (float*)alloc((size_t)512 * 240 * 4);
  float* seas   = (float*)alloc((size_t)512 * 252 * 4);
  float* lpart  = (float*)alloc((size_t)512 * 4);
  unsigned short* x0 = (unsigned short*)alloc((size_t)217 * 512 * 32 * 2);
  const size_t ACT = (size_t)217 * 512 * 512;
  unsigned short* actA = (unsigned short*)alloc(ACT * 2);
  unsigned short* actB = (unsigned short*)alloc(ACT * 2);
  unsigned short* actC = (unsigned short*)alloc(ACT * 2);
  unsigned short* hst  = (unsigned short*)alloc((size_t)4096 * 512 * 2);
  float* cst   = (float*)alloc((size_t)4096 * 512 * 4);
  unsigned short* pxw[4]; unsigned short* phw[4];
  pxw[0] = (unsigned short*)alloc((size_t)1 * 128 * 512 * 2);    // K=30 -> 1 tile
  phw[0] = (unsigned short*)alloc((size_t)16 * 128 * 512 * 2);
  for (int l = 1; l < 4; ++l) {
    pxw[l] = (unsigned short*)alloc((size_t)16 * 128 * 512 * 2);
    phw[l] = (unsigned short*)alloc((size_t)16 * 128 * 512 * 2);
  }
  unsigned short* pnl = (unsigned short*)alloc((size_t)16 * 32 * 512 * 2);
  float* ofull = (float*)alloc((size_t)217 * 512 * 12 * 4);

  // ---- ES prep + data marshalling ----
  es_prep<<<2, 256, 0, stream>>>(train, lvc, ssc, sp, idxs, levels, seas, lpart);
  loss_reduce<<<1, 512, 0, stream>>>(lpart, out + LOSS);
  build_inputs<<<(217 * 512 * 32) / 256, 256, 0, stream>>>(train, seas, levels, info, x0);
  network_act_k<<<(205 * 512 * 12 + 255) / 256, 256, 0, stream>>>(train, seas, levels, out + NA);
  holdout_k<<<(512 * 12 + 255) / 256, 256, 0, stream>>>(val, seas, levels, out + HA, out + HADN);

  // ---- weight packs (WMMA B-operand tiling, bf16) ----
  pack_w<<<(1 * 128 * 512) / 256, 256, 0, stream>>>(Wih[0], pxw[0], 2048, 30, 128, 1);
  pack_w<<<(16 * 128 * 512) / 256, 256, 0, stream>>>(Whh[0], phw[0], 2048, 512, 128, 16);
  for (int l = 1; l < 4; ++l) {
    pack_w<<<(16 * 128 * 512) / 256, 256, 0, stream>>>(Wih[l], pxw[l], 2048, 512, 128, 16);
    pack_w<<<(16 * 128 * 512) / 256, 256, 0, stream>>>(Whh[l], phw[l], 2048, 512, 128, 16);
  }
  pack_w<<<(16 * 32 * 512) / 256, 256, 0, stream>>>(nlW, pnl, 512, 512, 32, 16);

  // ---- dilated LSTM stack (compute once on full 217 steps; causality) ----
  auto run_layer = [&](const unsigned short* xs, int ldx, int kxt,
                       const unsigned short* wxp, const unsigned short* whp,
                       const float* bias, unsigned short* aout, int d) {
    int R = d * 512;
    int steps = (217 + d - 1) / d;     // Tp/d
    hipMemsetAsync(hst, 0, (size_t)R * 512 * 2, stream);
    hipMemsetAsync(cst, 0, (size_t)R * 512 * 4, stream);
    dim3 g(R / 16, 4);                 // 8 waves x 4 y-blocks = 32 hc-tiles
    for (int k = 0; k < steps; ++k) {
      lstm_step_fused<<<g, 256, 0, stream>>>(xs, ldx, kxt, wxp, whp, bias,
                                             cst, hst, aout, k, d);
    }
  };

  run_layer(x0,   32,  1, pxw[0], phw[0], bi[0], actA, 1);   // block0 L0
  run_layer(actA, 512, 16, pxw[1], phw[1], bi[1], actB, 2);  // block0 L1 (res src)
  run_layer(actB, 512, 16, pxw[2], phw[2], bi[2], actC, 4);  // block1 L2
  run_layer(actC, 512, 16, pxw[3], phw[3], bi[3], actA, 8);  // block1 L3
  resadd<<<(unsigned)((ACT + 255) / 256), 256, 0, stream>>>(actA, actB, actC, ACT);

  // ---- head: tanh(h @ nl_W^T + nl_b) then @ sc_W^T + sc_b ----
  head_gemm<<<dim3(6944, 1), 256, 0, stream>>>(actA, actC, pnl, nlb);
  sc_head<<<(217 * 512 * 12 + 255) / 256, 256, 0, stream>>>(actA, scW, scb, ofull);

  // ---- scatter outputs ----
  finalize_k<<<(217 * 512 * 12 + 255) / 256, 256, 0, stream>>>(
      ofull, seas, levels, out + PRED, out + ONT, out + HOP);
}
